// CUTLASSGroupedLinear_23003844838134
// MI455X (gfx1250) — matrix operand order repaired and checked by reading, not compiled, not run on hardware
//
#include <hip/hip_runtime.h>
#include <hip/hip_bf16.h>

// ---------------------------------------------------------------------------
// Grouped GEMM  Y_e = X_e @ W_e^T  (fp32 in/out) for MI455X (gfx1250).
//
// Strategy: bf16x3 split-precision on the CDNA5 matrix cores.
//   x = hi + lo (both bf16);  x*w ~= hi_x*hi_w + hi_x*lo_w + lo_x*hi_w
// accumulated in f32 via v_wmma_f32_16x16x32_bf16. This is ~2.7x the
// throughput of the native V_WMMA_F32_16X16X4_F32 path at ~fp32 accuracy.
//
// Tiling: 128x128 output tile / block, BK=32, 256 threads = 8 wave32 waves
// arranged 4(M) x 2(N); each wave computes a 32x64 sub-tile = 2x4 WMMA tiles.
// Expert routing is uniform per 128-row block (all M_SIZES are multiples of
// 512, so tiles never straddle an expert boundary).
// ---------------------------------------------------------------------------

typedef __attribute__((ext_vector_type(16))) __bf16 v16bf;
typedef __attribute__((ext_vector_type(8)))  float  v8f;

#define IN_F   2048
#define OUT_F  4096
#define NEXP   8
#define BM     128
#define BN     128
#define BK     32
#define LDA    40              // padded bf16 row stride (32 + 8) -> conflict-free-ish
#define NKSTEP (IN_F / BK)     // 64

union FragBF {                 // one WMMA operand (16 bf16 per lane = 32B)
    v16bf v;
    uint4 q[2];
};
union Pack4 {                  // 4 bf16 packed for an 8-byte LDS store
    __bf16 h[4];
    uint2  u;
};

__global__ __launch_bounds__(256)
void grouped_gemm_bf16x3(const float* __restrict__ X,
                         const float* __restrict__ W,
                         const long long* __restrict__ m_sizes,
                         const long long* __restrict__ m_offsets,
                         float* __restrict__ Y)
{
    (void)m_sizes;  // ranges fully defined by m_offsets

    __shared__ __align__(16) __bf16 sAhi[BM][LDA];
    __shared__ __align__(16) __bf16 sAlo[BM][LDA];
    __shared__ __align__(16) __bf16 sBhi[BN][LDA];
    __shared__ __align__(16) __bf16 sBlo[BN][LDA];

    const int n0   = blockIdx.x * BN;   // output-feature tile
    const int row0 = blockIdx.y * BM;   // token tile (global, sorted by expert)

    // ---- expert routing: uniform per block (scalar path) ----
    int e = 0;
    for (int i = 0; i < NEXP; ++i) {
        long long lo = m_offsets[i], hi = m_offsets[i + 1];
        if ((long long)row0 >= lo && (long long)row0 < hi) e = i;
    }
    const float* __restrict__ Xb = X + (size_t)row0 * IN_F;
    const float* __restrict__ Wb = W + (size_t)e * OUT_F * IN_F + (size_t)n0 * IN_F;

    const int tid   = threadIdx.x;
    const int lane  = tid & 31;
    const int wave  = tid >> 5;
    const int waveM = wave & 3;    // 0..3 -> 32-row group
    const int waveN = wave >> 2;   // 0..1 -> 64-col group
    const int half  = lane >> 4;   // K-half selector per ISA layout
    const int l16   = lane & 15;

    // ---- staging registers for the next K-slab (4 x float4 each) ----
    float4 ra[4], rb[4];

    auto load_tile = [&](int k0) {
#pragma unroll
        for (int i = 0; i < 4; ++i) {
            int idx = i * 256 + tid;          // 1024 float4 covers 128x32 f32
            int r   = idx >> 3;               // row 0..127
            int kq  = (idx & 7) << 2;         // k sub-offset 0,4,...,28
            ra[i] = *(const float4*)(Xb + (size_t)r * IN_F + k0 + kq);
            rb[i] = *(const float4*)(Wb + (size_t)r * IN_F + k0 + kq);
        }
    };

    auto stage_to_lds = [&]() {
#pragma unroll
        for (int i = 0; i < 4; ++i) {
            int idx = i * 256 + tid;
            int r   = idx >> 3;
            int kq  = (idx & 7) << 2;
            float av[4] = {ra[i].x, ra[i].y, ra[i].z, ra[i].w};
            float bv[4] = {rb[i].x, rb[i].y, rb[i].z, rb[i].w};
            Pack4 ah, al, bh, bl;
#pragma unroll
            for (int j = 0; j < 4; ++j) {
                __bf16 h = (__bf16)av[j];
                ah.h[j] = h;
                al.h[j] = (__bf16)(av[j] - (float)h);
                __bf16 g = (__bf16)bv[j];
                bh.h[j] = g;
                bl.h[j] = (__bf16)(bv[j] - (float)g);
            }
            *(uint2*)&sAhi[r][kq] = ah.u;      // ds_store_b64
            *(uint2*)&sAlo[r][kq] = al.u;
            *(uint2*)&sBhi[r][kq] = bh.u;
            *(uint2*)&sBlo[r][kq] = bl.u;
        }
    };

    v8f acc[2][4] = {};   // f32 accumulators, 64 VGPRs

    load_tile(0);

    for (int ks = 0; ks < NKSTEP; ++ks) {
        stage_to_lds();
        __syncthreads();

        // issue next slab's global loads so they overlap the WMMA work
        if (ks + 1 < NKSTEP) load_tile((ks + 1) * BK);

        // warm L2 for slab ks+2 (global_prefetch_b8)
        if (ks + 2 < NKSTEP) {
            const int k2 = (ks + 2) * BK;
            const int pr = tid & 127;
            const int po = (tid >> 7) * 16;   // both 64B halves of the row slab
            __builtin_prefetch(Xb + (size_t)pr * IN_F + k2 + po, 0, 3);
            __builtin_prefetch(Wb + (size_t)pr * IN_F + k2 + po, 0, 3);
        }

        // ---- A fragments for this wave's two 16-row groups (hi + lo) ----
        // ISA A-layout (16-bit, 16x32): lane L holds row (L&15);
        // K = 8*(L>>4) + {0..7} in q[0], and +16 more in q[1].
        FragBF ahi[2], alo[2];
#pragma unroll
        for (int mi = 0; mi < 2; ++mi) {
            int r = waveM * 32 + mi * 16 + l16;
            ahi[mi].q[0] = *(const uint4*)&sAhi[r][half * 8];
            ahi[mi].q[1] = *(const uint4*)&sAhi[r][half * 8 + 16];
            alo[mi].q[0] = *(const uint4*)&sAlo[r][half * 8];
            alo[mi].q[1] = *(const uint4*)&sAlo[r][half * 8 + 16];
        }

        // ---- per N-fragment: load B (hi+lo), run 3 WMMAs per acc tile ----
        // ISA B-layout (16-bit, 32x16): lane L holds col (L&15);
        // K = 8*(L>>4) + {0..7} in q[0], and +16 more in q[1] (matches A).
#pragma unroll
        for (int ni = 0; ni < 4; ++ni) {
            int n = waveN * 64 + ni * 16 + l16;
            FragBF bhi, blo;
            bhi.q[0] = *(const uint4*)&sBhi[n][half * 8];
            bhi.q[1] = *(const uint4*)&sBhi[n][half * 8 + 16];
            blo.q[0] = *(const uint4*)&sBlo[n][half * 8];
            blo.q[1] = *(const uint4*)&sBlo[n][half * 8 + 16];
#pragma unroll
            for (int mi = 0; mi < 2; ++mi) {
                acc[mi][ni] = __builtin_amdgcn_wmma_f32_16x16x32_bf16(
                    false, ahi[mi].v, false, bhi.v, (short)0, acc[mi][ni], false, false);
                acc[mi][ni] = __builtin_amdgcn_wmma_f32_16x16x32_bf16(
                    false, ahi[mi].v, false, blo.v, (short)0, acc[mi][ni], false, false);
                acc[mi][ni] = __builtin_amdgcn_wmma_f32_16x16x32_bf16(
                    false, alo[mi].v, false, bhi.v, (short)0, acc[mi][ni], false, false);
            }
        }
        __syncthreads();
    }

    // ---- epilogue: C/D layout -> lane (L&15)=col, VGPR v -> row v+8*(L>>4) ----
    float* __restrict__ Yb = Y + (size_t)row0 * OUT_F + n0;
#pragma unroll
    for (int mi = 0; mi < 2; ++mi) {
#pragma unroll
        for (int ni = 0; ni < 4; ++ni) {
            int mbase = waveM * 32 + mi * 16 + half * 8;
            int n     = waveN * 64 + ni * 16 + l16;
            float* po = Yb + (size_t)mbase * OUT_F + n;
#pragma unroll
            for (int v = 0; v < 8; ++v)
                po[(size_t)v * OUT_F] = acc[mi][ni][v];
        }
    }
}

extern "C" void kernel_launch(void* const* d_in, const int* in_sizes, int n_in,
                              void* d_out, int out_size, void* d_ws, size_t ws_size,
                              hipStream_t stream) {
    (void)n_in; (void)out_size; (void)d_ws; (void)ws_size;
    const float*     X        = (const float*)d_in[0];
    const float*     W        = (const float*)d_in[1];
    const long long* msz      = (const long long*)d_in[2];
    const long long* moff     = (const long long*)d_in[3];
    float*           Y        = (float*)d_out;

    const int tokens = in_sizes[0] / IN_F;            // 16384
    dim3 grid(OUT_F / BN, (tokens + BM - 1) / BM);    // 32 x 128 = 4096 blocks
    dim3 block(256);
    hipLaunchKernelGGL(grouped_gemm_bf16x3, grid, block, 0, stream,
                       X, W, msz, moff, Y);
}
